// GRU_90821378441798
// MI455X (gfx1250) — compile-verified, matrix operand order checked by reading, not run on hardware
//
#include <hip/hip_runtime.h>
#include <hip/hip_bf16.h>

typedef __attribute__((ext_vector_type(16))) __bf16 v16bf;
typedef __attribute__((ext_vector_type(8)))  float  v8f;

#define BDIM 4
#define NDIM 64
#define LDIM 1024
#define FDIM 15
#define CDIM 128
#define FB_STRIDE 136   // padded bf16 row stride for feats (kills LDS bank conflicts)
#define HG_STRIDE 264   // padded f32 row stride for hg

// ---------------------------------------------------------------------------
// Pre-kernel: swizzle W_gru (f32, [2][128][256]) into bf16 WMMA-B fragment
// layout: flat idx = ((((layer*4+kt)*16+nt)*32+lane)*16+i); lane holds col
// n = nt*16 + lane%16; element i holds k = kt*32 + (i/8)*16 + (lane/16)*8 + i%8
// ---------------------------------------------------------------------------
__global__ __launch_bounds__(256) void swizzle_wg(const float* __restrict__ Wg,
                                                  __hip_bfloat16* __restrict__ Bsw) {
    int idx   = blockIdx.x * 256 + threadIdx.x;     // 0..65535
    int i     = idx & 15;
    int lane  = (idx >> 4) & 31;
    int nt    = (idx >> 9) & 15;
    int kt    = (idx >> 13) & 3;
    int layer = idx >> 15;
    int k = kt * 32 + ((i >> 3) << 4) + ((lane >> 4) << 3) + (i & 7);
    int n = nt * 16 + (lane & 15);
    Bsw[idx] = __float2bfloat16(Wg[layer * (CDIM * 2 * CDIM) + k * (2 * CDIM) + n]);
}

// ---------------------------------------------------------------------------
// Fused kernel: one workgroup per (b, l). 256 threads = 8 waves.
// ---------------------------------------------------------------------------
__global__ __launch_bounds__(256) void gru_fused(
    const float* __restrict__ xs, const float* __restrict__ x,
    const float* __restrict__ W1, const float* __restrict__ b1,
    const float* __restrict__ Wout, const float* __restrict__ bout,
    const __hip_bfloat16* __restrict__ Bsw, float* __restrict__ out)
{
    __shared__ __align__(16) __hip_bfloat16 s_featb[NDIM][FB_STRIDE]; // ~17 KB
    __shared__ __align__(16) float s_hg[NDIM][HG_STRIDE];             // ~66 KB
    __shared__ float s_W1[16][CDIM];                                  // 8 KB
    __shared__ float s_b1[CDIM];
    __shared__ float s_Wo[CDIM];
    __shared__ float s_xl[NDIM];
    __shared__ float s_xs[NDIM][16];                                  // 4 KB

    const int tid = threadIdx.x;
    const int bl  = blockIdx.x;
    const int b   = bl >> 10;
    const int l   = bl & (LDIM - 1);

    // ---- cooperative loads -------------------------------------------------
    for (int i = tid; i < 16 * CDIM; i += 256) s_W1[i >> 7][i & 127] = W1[i];
    if (tid < CDIM) { s_b1[tid] = b1[tid]; s_Wo[tid] = Wout[tid]; }
    if (tid < NDIM) s_xl[tid] = x[(b * NDIM + tid) * LDIM + l];
    for (int i = tid; i < NDIM * FDIM; i += 256) {
        int n = i / FDIM, f = i - n * FDIM;
        s_xs[n][f] = xs[(b * NDIM + n) * FDIM + f];
    }
    __syncthreads();

    // ---- phase 1: feats = relu([x, xs] @ W1 + b1) -> bf16 ------------------
    for (int r = 0; r < 32; ++r) {
        int idx = tid + r * 256;
        int n = idx >> 7, c = idx & 127;
        float acc = s_b1[c] + s_xl[n] * s_W1[0][c];
        #pragma unroll
        for (int f = 0; f < FDIM; ++f) acc += s_xs[n][f] * s_W1[f + 1][c];
        s_featb[n][c] = __float2bfloat16(fmaxf(acc, 0.f));
    }
    __syncthreads();

    const int wave   = tid >> 5, lane = tid & 31;
    const int mtile  = wave & 3;            // 4 M-tiles of 16 rows
    const int ntbase = (wave >> 2) * 8;     // waves 0-3: nt 0..7, 4-7: nt 8..15
    const int lmod   = lane & 15, lgrp = lane >> 4;
    const int mrow   = mtile * 16 + lmod;   // A-fragment row for this lane

    for (int layer = 0; layer < 2; ++layer) {
        // ---- WMMA GEMM: hg(64x256) = feats(64x128) @ Wg(128x256) ----------
        union { v16bf v; unsigned u[8]; } A[4];
        const unsigned* featu = (const unsigned*)&s_featb[0][0];
        #pragma unroll
        for (int kt = 0; kt < 4; ++kt) {
            #pragma unroll
            for (int d = 0; d < 8; ++d) {
                int k = kt * 32 + ((d >> 2) << 4) + (lgrp << 3) + ((d & 3) << 1);
                A[kt].u[d] = featu[(mrow * FB_STRIDE + k) >> 1];
            }
        }
        #pragma unroll
        for (int nt8 = 0; nt8 < 8; ++nt8) {
            int nt = ntbase + nt8;
            v8f acc = {};
            #pragma unroll
            for (int kt = 0; kt < 4; ++kt) {
                union { v16bf v; uint4 q[2]; } Bm;
                const uint4* p = (const uint4*)(Bsw +
                    ((((layer * 4 + kt) * 16 + nt) * 32 + lane) << 4));
                Bm.q[0] = p[0];
                Bm.q[1] = p[1];
                acc = __builtin_amdgcn_wmma_f32_16x16x32_bf16(
                    false, A[kt].v, false, Bm.v, (short)0, acc, false, false);
            }
            int ng = nt * 16 + lmod;
            #pragma unroll
            for (int j = 0; j < 8; ++j)
                s_hg[mtile * 16 + lgrp * 8 + j][ng] = acc[j];
        }
        __syncthreads();

        // ---- scan: online logsumexp over N per channel --------------------
        if (tid < CDIM) {
            int c = tid;
            float a = 0.f, mrun = 0.f, srun = 0.f;
            #pragma unroll 1
            for (int n = 0; n < NDIM; ++n) {
                float h = s_hg[n][c];
                float g = s_hg[n][c + CDIM];
                // softplus(g) = max(g,0) + log1p(exp(-|g|))
                float sp = fmaxf(g, 0.f) + __logf(1.f + __expf(-fabsf(g)));
                // log_g(h): h>=0 -> log(h+0.5); h<0 -> -softplus(-h) = h - log1p(exp(h))
                float lgh = (h >= 0.f) ? __logf(h + 0.5f)
                                       : (h - __logf(1.f + __expf(h)));
                a += -sp;                         // a_star (cumsum of log_coeffs)
                float t = (g - sp) + lgh - a;     // log_values - a_star
                if (n == 0)        { mrun = t; srun = 1.f; }
                else if (t > mrun) { srun = srun * __expf(mrun - t) + 1.f; mrun = t; }
                else               { srun += __expf(t - mrun); }
                float val = __expf(a + mrun + __logf(srun));
                s_featb[n][c] = __float2bfloat16(val);
            }
        }
        __syncthreads();
    }

    // ---- output: feats @ W_out + b_out + x ---------------------------------
    if (tid < NDIM) {
        int n = tid;
        float acc = bout[0] + s_xl[n];
        #pragma unroll
        for (int c = 0; c < CDIM; ++c)
            acc += __bfloat162float(s_featb[n][c]) * s_Wo[c];
        out[(b * NDIM + n) * LDIM + l] = acc;
    }
}

// ---------------------------------------------------------------------------
extern "C" void kernel_launch(void* const* d_in, const int* in_sizes, int n_in,
                              void* d_out, int out_size, void* d_ws, size_t ws_size,
                              hipStream_t stream) {
    const float* xs  = (const float*)d_in[0];
    const float* x   = (const float*)d_in[1];
    const float* W1  = (const float*)d_in[2];
    const float* b1  = (const float*)d_in[3];
    const float* Wg  = (const float*)d_in[4];
    const float* Wo  = (const float*)d_in[5];
    const float* bo  = (const float*)d_in[6];
    float* out = (float*)d_out;

    __hip_bfloat16* Bsw = (__hip_bfloat16*)d_ws;   // 65536 bf16 = 128 KB

    swizzle_wg<<<256, 256, 0, stream>>>(Wg, Bsw);
    gru_fused<<<BDIM * LDIM, 256, 0, stream>>>(xs, x, W1, b1, Wo, bo, Bsw, out);
}
